// LSTMCell_3685081940484
// MI455X (gfx1250) — matrix operand order repaired and checked by reading, not compiled, not run on hardware
//
#include <hip/hip_runtime.h>
#include <hip/hip_bf16.h>
#include <stdint.h>

// ---------------------------------------------------------------------------
// Fused LSTM cell for MI455X (gfx1250, wave32, WMMA + async LDS staging).
//   gates = [x|h] @ [Wi;Wf;Wg;Wo]^T + b   -> sigmoid/tanh fusion -> h', c'
// Pass 1: fp32 -> bf16 conversion of activations (xh) and weights into d_ws.
// Pass 2: block = 8 waves, block tile 512(M) x 16(N) x 4 gates.
//         The block's full B tile (4*16 rows x K=2048 bf16 = 256 KB) is
//         staged ONCE into LDS with GLOBAL_LOAD_ASYNC_TO_LDS_B128 (XOR
//         swizzled), then the K loop runs with no further synchronization:
//         A fragments stream from L2, B fragments from LDS, 16 WMMAs/k-step
//         per wave, fused activation epilogue writes h_new / c_new.
// ---------------------------------------------------------------------------

typedef __attribute__((ext_vector_type(16))) __bf16 v16bf;
typedef __attribute__((ext_vector_type(8)))  __bf16 bf16x8;
typedef __attribute__((ext_vector_type(8)))  float  v8f;
typedef __attribute__((ext_vector_type(4)))  int    i32x4;

typedef __attribute__((address_space(1))) i32x4 gas_i32x4;  // global int4
typedef __attribute__((address_space(3))) i32x4 las_i32x4;  // LDS int4

#define LSTM_BATCH 16384
#define LSTM_IN    1024
#define LSTM_HID   1024
#define LSTM_K     2048   // IN + HID
#define LSTM_NGATE 4

#define BLOCK_M    512    // 8 waves * 64 rows
#define WAVE_M     64     // 4 sub-tiles of 16 rows per wave
#define LDS_BYTES  (64 * 256 * 16)   // 64 B-rows x 256 chunks x 16 B = 256 KB

// ---------------- conversion pass: fp32 -> bf16, vectorized 8-wide ----------

__device__ __forceinline__ bf16x8 pack8(const float* __restrict__ src) {
    float4 f0 = ((const float4*)src)[0];
    float4 f1 = ((const float4*)src)[1];
    bf16x8 o;
    o[0] = (__bf16)f0.x; o[1] = (__bf16)f0.y; o[2] = (__bf16)f0.z; o[3] = (__bf16)f0.w;
    o[4] = (__bf16)f1.x; o[5] = (__bf16)f1.y; o[6] = (__bf16)f1.z; o[7] = (__bf16)f1.w;
    return o;
}

__global__ void lstm_cvt_xh_kernel(const float* __restrict__ x,
                                   const float* __restrict__ h,
                                   __bf16* __restrict__ xh) {
    const size_t nchunk = (size_t)LSTM_BATCH * LSTM_K / 8;
    for (size_t t = blockIdx.x * (size_t)blockDim.x + threadIdx.x; t < nchunk;
         t += (size_t)gridDim.x * blockDim.x) {
        size_t e = t * 8;
        size_t b = e >> 11;              // / LSTM_K
        int    i = (int)(e & (LSTM_K - 1));
        const float* src = (i < LSTM_IN) ? (x + b * LSTM_IN + i)
                                         : (h + b * LSTM_HID + (i - LSTM_IN));
        *(bf16x8*)(xh + e) = pack8(src);
    }
}

__global__ void lstm_cvt_w_kernel(const float* __restrict__ wi,
                                  const float* __restrict__ wf,
                                  const float* __restrict__ wg,
                                  const float* __restrict__ wo,
                                  __bf16* __restrict__ wc) {
    const size_t per_gate = (size_t)LSTM_HID * LSTM_K;   // 2M elements
    const size_t nchunk   = (size_t)LSTM_NGATE * per_gate / 8;
    for (size_t t = blockIdx.x * (size_t)blockDim.x + threadIdx.x; t < nchunk;
         t += (size_t)gridDim.x * blockDim.x) {
        size_t e = t * 8;
        int    g = (int)(e / per_gate);
        size_t r = e - (size_t)g * per_gate;
        const float* base = (g == 0) ? wi : (g == 1) ? wf : (g == 2) ? wg : wo;
        *(bf16x8*)(wc + e) = pack8(base + r);
    }
}

// ---------------- WMMA GEMM + fused LSTM epilogue ---------------------------

__device__ __forceinline__ v16bf cat8(bf16x8 lo, bf16x8 hi) {
    return __builtin_shufflevector(lo, hi, 0, 1, 2, 3, 4, 5, 6, 7,
                                           8, 9, 10, 11, 12, 13, 14, 15);
}

__device__ __forceinline__ float fast_sigmoid(float v) {
    return 1.0f / (1.0f + __expf(-v));
}
__device__ __forceinline__ float fast_tanh(float v) {
    float e = __expf(2.0f * v);
    return (e - 1.0f) / (e + 1.0f);
}

__global__ __launch_bounds__(256)
void lstm_wmma_kernel(const __bf16* __restrict__ xh,   // [BATCH, K] bf16
                      const __bf16* __restrict__ wc,   // [4*HID, K] bf16
                      const float* __restrict__ bi, const float* __restrict__ bfv,
                      const float* __restrict__ bg, const float* __restrict__ bo,
                      const float* __restrict__ c_old, // [BATCH, HID] f32
                      float* __restrict__ h_new,       // [BATCH, HID] f32
                      float* __restrict__ c_new) {     // [BATCH, HID] f32
    extern __shared__ char smem[];   // 256 KB B tile, XOR-swizzled

    const int tid   = threadIdx.x;
    const int lane  = tid & 31;
    const int wave  = tid >> 5;
    const int nidx  = lane & 15;   // output column / A row within 16-tile
    const int khalf = lane >> 4;   // which K half this lane carries
    const int j0 = blockIdx.y * 16;                    // hidden tile base
    const int m0 = blockIdx.x * BLOCK_M + wave * WAVE_M;

    // ---- stage the block's B tile into LDS (once; ASYNCcnt tracked) ----
    // LDS row r = gate*16 + n  <=>  weight row gate*HID + j0 + n.
    // Row = 256 16-byte chunks; chunk index XOR (r & 15) so the 16-lane
    // column reads in the K loop land on 16 distinct bank groups.
    for (int s = 0; s < 64; ++s) {
        const int g = s >> 4, n = s & 15;
        const __bf16* gsrc =
            wc + (size_t)(g * LSTM_HID + j0 + n) * LSTM_K + (size_t)tid * 8;
        const unsigned loff =
            (((unsigned)s << 8) + ((unsigned)tid ^ (unsigned)(s & 15))) << 4;
#if __has_builtin(__builtin_amdgcn_global_load_async_to_lds_b128)
        __builtin_amdgcn_global_load_async_to_lds_b128(
            (gas_i32x4*)gsrc, (las_i32x4*)(smem + loff), 0, 0);
#else
        *(bf16x8*)(smem + loff) = *(const bf16x8*)gsrc;   // fallback: vmem+ds
#endif
    }
#if __has_builtin(__builtin_amdgcn_s_wait_asynccnt)
    __builtin_amdgcn_s_wait_asynccnt(0);
#endif
    __syncthreads();   // all waves' staging complete -> B tile valid

    // ---- K loop: 16 WMMAs per step (4 M sub-tiles x 4 gates) ----
    v8f acc[4][4];
#pragma unroll
    for (int s = 0; s < 4; ++s)
#pragma unroll
        for (int g = 0; g < 4; ++g) acc[s][g] = (v8f){};

    // A: lane holds row (m0 + s*16 + nidx); two 8-elem chunks per k-step.
    const __bf16* arow0 = xh + (size_t)(m0 + nidx) * LSTM_K + khalf * 16;

    for (int k = 0; k < LSTM_K; k += 32) {
        const int c0 = (k >> 3) + khalf * 2;   // 16B chunk index in B row
        v16bf Bfr[4];
#pragma unroll
        for (int g = 0; g < 4; ++g) {
            const unsigned base = (unsigned)((g * 16 + nidx) << 8);
            const unsigned o0 = (base + ((unsigned)c0       ^ (unsigned)nidx)) << 4;
            const unsigned o1 = (base + ((unsigned)(c0 + 1) ^ (unsigned)nidx)) << 4;
            Bfr[g] = cat8(*(const bf16x8*)(smem + o0),
                          *(const bf16x8*)(smem + o1));
        }
#pragma unroll
        for (int s = 0; s < 4; ++s) {
            const __bf16* ap = arow0 + (size_t)s * 16 * LSTM_K + k;
            v16bf a = cat8(*(const bf16x8*)ap, *(const bf16x8*)(ap + 8));
            acc[s][0] = __builtin_amdgcn_wmma_f32_16x16x32_bf16(
                false, a, false, Bfr[0], (short)0, acc[s][0], false, false);
            acc[s][1] = __builtin_amdgcn_wmma_f32_16x16x32_bf16(
                false, a, false, Bfr[1], (short)0, acc[s][1], false, false);
            acc[s][2] = __builtin_amdgcn_wmma_f32_16x16x32_bf16(
                false, a, false, Bfr[2], (short)0, acc[s][2], false, false);
            acc[s][3] = __builtin_amdgcn_wmma_f32_16x16x32_bf16(
                false, a, false, Bfr[3], (short)0, acc[s][3], false, false);
        }
    }

    // ---- fused epilogue ----
    // C/D layout: lane (khalf,nidx), VGPR r -> (M = khalf*8 + r, N = nidx).
    const int col = j0 + nidx;
    const float bia = bi[col];
    const float bfa = bfv[col];
    const float bga = bg[col];
    const float boa = bo[col];

#pragma unroll
    for (int s = 0; s < 4; ++s) {
        const int mbase = m0 + s * 16 + khalf * 8;
#pragma unroll
        for (int r = 0; r < 8; ++r) {
            const size_t idx = (size_t)(mbase + r) * LSTM_HID + col;
            float gi = fast_sigmoid(acc[s][0][r] + bia);
            float gf = fast_sigmoid(acc[s][1][r] + bfa);
            float gg = fast_tanh   (acc[s][2][r] + bga);
            float go = fast_sigmoid(acc[s][3][r] + boa);
            float cn = gf * c_old[idx] + gi * gg;
            c_new[idx] = cn;
            h_new[idx] = go * fast_tanh(cn);
        }
    }
}

// ---------------- launcher --------------------------------------------------

extern "C" void kernel_launch(void* const* d_in, const int* in_sizes, int n_in,
                              void* d_out, int out_size, void* d_ws, size_t ws_size,
                              hipStream_t stream) {
    (void)in_sizes; (void)n_in; (void)out_size; (void)ws_size;

    const float* x  = (const float*)d_in[0];
    const float* h  = (const float*)d_in[1];
    const float* c  = (const float*)d_in[2];
    const float* Wi = (const float*)d_in[3];
    const float* bi = (const float*)d_in[4];
    const float* Wf = (const float*)d_in[5];
    const float* bf = (const float*)d_in[6];
    const float* Wg = (const float*)d_in[7];
    const float* bg = (const float*)d_in[8];
    const float* Wo = (const float*)d_in[9];
    const float* bo = (const float*)d_in[10];

    float* h_new = (float*)d_out;
    float* c_new = h_new + (size_t)LSTM_BATCH * LSTM_HID;

    // Workspace: xh bf16 (64 MB) + concatenated weights bf16 (16 MB).
    __bf16* xh = (__bf16*)d_ws;
    __bf16* wc = xh + (size_t)LSTM_BATCH * LSTM_K;

    lstm_cvt_xh_kernel<<<2048, 256, 0, stream>>>(x, h, xh);
    lstm_cvt_w_kernel<<<1024, 256, 0, stream>>>(Wi, Wf, Wg, Wo, wc);

    dim3 grid(LSTM_BATCH / BLOCK_M, LSTM_HID / 16);
    lstm_wmma_kernel<<<grid, 256, LDS_BYTES, stream>>>(xh, wc, bi, bf, bg, bo,
                                                       c, h_new, c_new);
}
